// DynamicAttention_40827959116532
// MI455X (gfx1250) — matrix operand order repaired and checked by reading, not compiled, not run on hardware
//
#include <hip/hip_runtime.h>
#include <hip/hip_bf16.h>

// DynamicAttention for MI455X (gfx1250, wave32, WMMA + async-to-LDS).
// All GEMMs run on v_wmma_f32_16x16x32_bf16 (bf16 in, f32 accumulate).
//
// Pipeline:
//   1) proj:  qF = query@Wq+bq (f32), sB = bf16(src@Ws+bs), tB = bf16(trg@Ws+bs)
//   2) score: per (b,l): P = (diag(q) * skey) @ tkey^T / sqrt(512); fused softmax;
//             emit only row/col marginals ws[b,l,s], wt[b,l,t] (bf16).
//             Double-buffered LDS; tkey tile staged with global_load_async_to_lds.
//   3) ctx:   s_ctx[b] = ws[b] @ src[b], t_ctx[b] = wt[b] @ trg[b]  (bf16 out)
//   4) final: out = [query | s_ctx | t_ctx] @ Wo + bo  (K = 1536, f32 out)

typedef __attribute__((ext_vector_type(16))) __bf16 v16bf;
typedef __attribute__((ext_vector_type(8)))  float  v8f;

union Frag16 { v16bf v; unsigned int u[8]; };

static constexpr int LDK = 34;   // padded K-stride (bf16 elems) for LDS tiles

__device__ __forceinline__ v8f wmma_bf16(v16bf a, v16bf b, v8f c) {
  // (neg_a, A, neg_b, B, c_mod, C, reuse_a, reuse_b)
  return __builtin_amdgcn_wmma_f32_16x16x32_bf16(false, a, false, b, (short)0, c,
                                                 false, false);
}

// Low 32 bits of a generic pointer to __shared__ == wave-relative LDS byte addr.
__device__ __forceinline__ unsigned lds_addr_of(const void* p) {
  return (unsigned)(unsigned long long)p;
}

// Async copy of one [128 rows x 32 bf16] tile (global, row stride `gstride`
// bf16 elems, starting column kt) into LDS laid out [r][k] with stride LDK.
// 2048 B32 words / 256 threads = 8 issues per thread. ASYNCcnt-tracked.
__device__ __forceinline__ void async_tile_bf16(const __bf16* gbase, int gstride,
                                                int kt, const __bf16* ldsbase,
                                                int tid) {
  unsigned lb = lds_addr_of(ldsbase);
#pragma unroll
  for (int i = 0; i < 8; ++i) {
    int idx = tid + i * 256;
    int r = idx >> 4, w = idx & 15;              // word w covers k = 2w, 2w+1
    unsigned ldsa = lb + (unsigned)(r * (LDK * 2) + w * 4);
    unsigned voff = (unsigned)((r * gstride + kt) * 2 + w * 4);
    asm volatile("global_load_async_to_lds_b32 %0, %1, %2"
                 :: "v"(ldsa), "v"(voff), "s"(gbase) : "memory");
  }
}

__device__ __forceinline__ void wait_async0() {
  asm volatile("s_wait_asynccnt 0" ::: "memory");
}

// A fragment: 16x32 bf16, ISA layout:
//  lanes 0-15: M=lane, K = 0..7 (elems 0-7) and 16..23 (elems 8-15)
//  lanes16-31: M=lane-16, K = 8..15 and 24..31
__device__ __forceinline__ v16bf load_a_frag(const __bf16* lA, int lane) {
  int m  = lane & 15;
  int kb = (lane >> 4) * 8;
  const unsigned int* r32 = (const unsigned int*)(lA + m * LDK);
  Frag16 f;
#pragma unroll
  for (int j = 0; j < 4; ++j) {
    f.u[j]     = r32[(kb >> 1) + j];            // K = kb+2j, kb+2j+1
    f.u[4 + j] = r32[((kb + 16) >> 1) + j];     // K = kb+16+2j, ...
  }
  return f.v;
}

// B fragment: 32x16 bf16 (K x N), lB stored [n][k] with stride LDK:
//  lanes 0-15: N=lane, K = 0..7 and 16..23 ; lanes 16-31: N=lane-16, K = 8..15 and 24..31
__device__ __forceinline__ v16bf load_b_frag(const __bf16* lB, int lane) {
  int n  = lane & 15;
  int kb = (lane >> 4) * 8;
  const unsigned int* r32 = (const unsigned int*)(lB + n * LDK);
  Frag16 f;
#pragma unroll
  for (int j = 0; j < 4; ++j) {
    f.u[j]     = r32[(kb >> 1) + j];
    f.u[4 + j] = r32[((kb + 16) >> 1) + j];
  }
  return f.v;
}

// One 128x128x32 block step: wave handles rows 16*wave..16*wave+15, all 128 cols.
__device__ __forceinline__ void gemm_step(const __bf16* lA, const __bf16* lB,
                                          int lane, int wave, v8f acc[8]) {
  v16bf a = load_a_frag(lA + wave * 16 * LDK, lane);
#pragma unroll
  for (int f = 0; f < 8; ++f) {
    v16bf b = load_b_frag(lB + f * 16 * LDK, lane);
    acc[f] = wmma_bf16(a, b, acc[f]);
  }
}

// ---------------------------------------------------------------- projections
__global__ __launch_bounds__(256)
void da_proj_kernel(const float* __restrict__ A, const float* __restrict__ W,
                    const float* __restrict__ bias,
                    float* __restrict__ outF, __bf16* __restrict__ outB,
                    int M, int N, int K) {
  __shared__ __align__(16) __bf16 lA[128 * LDK];
  __shared__ __align__(16) __bf16 lB[128 * LDK];
  const int m0 = blockIdx.x * 128, n0 = blockIdx.y * 128;
  const int tid = threadIdx.x, lane = tid & 31, wave = tid >> 5;

  v8f acc[8];
#pragma unroll
  for (int f = 0; f < 8; ++f)
#pragma unroll
    for (int v = 0; v < 8; ++v) acc[f][v] = 0.0f;

  for (int kt = 0; kt < K; kt += 32) {
    for (int idx = tid; idx < 4096; idx += 256) {       // A: [m][k], k contig
      int k = idx & 31, m = idx >> 5;
      lA[m * LDK + k] = (__bf16)A[(size_t)(m0 + m) * K + kt + k];
    }
    for (int idx = tid; idx < 4096; idx += 256) {       // W row-major -> lB[n][k]
      int n = idx & 127, k = idx >> 7;
      lB[n * LDK + k] = (__bf16)W[(size_t)(kt + k) * N + n0 + n];
    }
    __syncthreads();
    gemm_step(lA, lB, lane, wave, acc);
    __syncthreads();
  }

  const int half = lane >> 4, nl = lane & 15;
#pragma unroll
  for (int f = 0; f < 8; ++f)
#pragma unroll
    for (int v = 0; v < 8; ++v) {
      int row = m0 + wave * 16 + v + 8 * half;
      int col = n0 + f * 16 + nl;
      float val = acc[f][v] + bias[col];
      if (outF) outF[(size_t)row * N + col] = val;
      if (outB) outB[(size_t)row * N + col] = (__bf16)val;
    }
}

// ------------------------------------------------- trilinear score + softmax
__global__ __launch_bounds__(256)
void da_score_kernel(const float*  __restrict__ qF,   // [2048][512] f32
                     const __bf16* __restrict__ sB,   // [2048][512]
                     const __bf16* __restrict__ tB,   // [2048][512]
                     __bf16* __restrict__ wsOut,      // [2048][128]
                     __bf16* __restrict__ wtOut) {    // [2048][128]
  __shared__ __align__(16) float  qrow[512];
  __shared__ __align__(16) __bf16 lA[2][128 * LDK];   // double-buffered
  __shared__ __align__(16) __bf16 lB[2][128 * LDK];
  __shared__ float rowsum[128];
  __shared__ float colsum[128];
  __shared__ float red[8];
  __shared__ float bcast[2];

  const int tid = threadIdx.x, lane = tid & 31, wave = tid >> 5;
  const int bl = blockIdx.x;          // b*128 + l
  const int b  = bl >> 7;

  const __bf16* sBase = sB + (size_t)b * 128 * 512;
  const __bf16* tBase = tB + (size_t)b * 128 * 512;

  for (int i = tid; i < 512; i += 256) qrow[i] = qF[(size_t)bl * 512 + i];
  if (tid < 128) colsum[tid] = 0.0f;
  __syncthreads();

  v8f acc[8];
#pragma unroll
  for (int f = 0; f < 8; ++f)
#pragma unroll
    for (int v = 0; v < 8; ++v) acc[f][v] = 0.0f;

  // Tile loader: q-scaled skey (VALU, needs fp32 scale+convert) + async tkey.
  auto load_tiles = [&](int kt, int buf) {
    async_tile_bf16(tBase, 512, kt, &lB[buf][0], tid);   // ASYNCcnt path
    for (int idx = tid; idx < 4096; idx += 256) {
      int k = idx & 31, r = idx >> 5;
      float sv = (float)sBase[(size_t)r * 512 + kt + k];
      lA[buf][r * LDK + k] = (__bf16)(sv * qrow[kt + k]);
    }
  };

  int cur = 0;
  load_tiles(0, 0);
  wait_async0();
  __syncthreads();

  for (int kt = 32; kt < 512; kt += 32) {
    load_tiles(kt, cur ^ 1);                 // prefetch next tile
    gemm_step(&lA[cur][0], &lB[cur][0], lane, wave, acc);
    wait_async0();
    __syncthreads();
    cur ^= 1;
  }
  gemm_step(&lA[cur][0], &lB[cur][0], lane, wave, acc);

  // ---- stable softmax over all 128x128 scores (distributed in accumulators)
  const float scale = 0.04419417382415922f;  // 1/sqrt(512)
  float lmax = -1e30f;
#pragma unroll
  for (int f = 0; f < 8; ++f)
#pragma unroll
    for (int v = 0; v < 8; ++v) {
      float x = acc[f][v] * scale;
      acc[f][v] = x;
      lmax = fmaxf(lmax, x);
    }
#pragma unroll
  for (int m = 1; m <= 16; m <<= 1) lmax = fmaxf(lmax, __shfl_xor(lmax, m, 32));
  if (lane == 0) red[wave] = lmax;
  __syncthreads();
  if (tid == 0) {
    float g = red[0];
#pragma unroll
    for (int i = 1; i < 8; ++i) g = fmaxf(g, red[i]);
    bcast[0] = g;
  }
  __syncthreads();
  const float gmax = bcast[0];

  float rs[8], cs[8];
#pragma unroll
  for (int i = 0; i < 8; ++i) { rs[i] = 0.0f; cs[i] = 0.0f; }
#pragma unroll
  for (int f = 0; f < 8; ++f)
#pragma unroll
    for (int v = 0; v < 8; ++v) {
      float e = __expf(acc[f][v] - gmax);
      rs[v] += e;        // sum over t (fragments)
      cs[f] += e;        // sum over s (rows in strip)
    }

  // Row sums: reduce over the 16 lanes of each half (same s, different t).
#pragma unroll
  for (int v = 0; v < 8; ++v)
#pragma unroll
    for (int m = 1; m <= 8; m <<= 1) rs[v] += __shfl_xor(rs[v], m, 32);
  if ((lane & 15) == 0) {
    int half = lane >> 4;
#pragma unroll
    for (int v = 0; v < 8; ++v)
      rowsum[wave * 16 + half * 8 + v] = rs[v];
  }

  // Col sums: fold the two halves (s and s+8), then accumulate across waves.
#pragma unroll
  for (int f = 0; f < 8; ++f) cs[f] += __shfl_xor(cs[f], 16, 32);
  if (lane < 16) {
#pragma unroll
    for (int f = 0; f < 8; ++f)
      atomicAdd(&colsum[f * 16 + lane], cs[f]);
  }
  __syncthreads();

  if (wave == 0) {
    float z = 0.0f;
#pragma unroll
    for (int i = 0; i < 4; ++i) z += rowsum[lane + 32 * i];
#pragma unroll
    for (int m = 1; m <= 16; m <<= 1) z += __shfl_xor(z, m, 32);
    if (lane == 0) bcast[1] = 1.0f / z;
  }
  __syncthreads();
  const float zinv = bcast[1];

  if (tid < 128) {
    wsOut[(size_t)bl * 128 + tid] = (__bf16)(rowsum[tid] * zinv);
  } else {
    int t = tid - 128;
    wtOut[(size_t)bl * 128 + t] = (__bf16)(colsum[t] * zinv);
  }
}

// --------------------------------------------------------- context GEMMs
__global__ __launch_bounds__(256)
void da_ctx_kernel(const __bf16* __restrict__ wsB, const __bf16* __restrict__ wtB,
                   const float*  __restrict__ src, const float* __restrict__ trg,
                   __bf16* __restrict__ scx, __bf16* __restrict__ tcx) {
  __shared__ __align__(16) __bf16 lA[128 * LDK];
  __shared__ __align__(16) __bf16 lB[128 * LDK];
  const int n0 = blockIdx.x * 128;
  const int b  = blockIdx.y;
  const int sel = blockIdx.z;          // 0 = src path, 1 = trg path
  const int tid = threadIdx.x, lane = tid & 31, wave = tid >> 5;

  const __bf16* Abase = (sel ? wtB : wsB) + (size_t)b * 128 * 128;  // [L][S]
  const float*  Bsrc  = (sel ? trg : src) + (size_t)b * 128 * 512;  // [S][512]
  __bf16*       outB  = (sel ? tcx : scx) + (size_t)b * 128 * 512;

  v8f acc[8];
#pragma unroll
  for (int f = 0; f < 8; ++f)
#pragma unroll
    for (int v = 0; v < 8; ++v) acc[f][v] = 0.0f;

  for (int kt = 0; kt < 128; kt += 32) {
    async_tile_bf16(Abase, 128, kt, lA, tid);            // bf16 weights tile
    for (int idx = tid; idx < 4096; idx += 256) {        // f32 -> bf16 convert
      int n = idx & 127, k = idx >> 7;
      lB[n * LDK + k] = (__bf16)Bsrc[(size_t)(kt + k) * 512 + n0 + n];
    }
    wait_async0();
    __syncthreads();
    gemm_step(lA, lB, lane, wave, acc);
    __syncthreads();
  }

  const int half = lane >> 4, nl = lane & 15;
#pragma unroll
  for (int f = 0; f < 8; ++f)
#pragma unroll
    for (int v = 0; v < 8; ++v) {
      int row = wave * 16 + v + 8 * half;
      int col = n0 + f * 16 + nl;
      outB[(size_t)row * 512 + col] = (__bf16)acc[f][v];
    }
}

// ----------------------------------------------------- final concat GEMM
__global__ __launch_bounds__(256)
void da_final_kernel(const float* __restrict__ query,      // [2048][512]
                     const __bf16* __restrict__ scx,       // [2048][512]
                     const __bf16* __restrict__ tcx,       // [2048][512]
                     const float* __restrict__ Wo,         // [1536][512]
                     const float* __restrict__ bo,         // [512]
                     float* __restrict__ out) {            // [2048][512]
  __shared__ __align__(16) __bf16 lA[128 * LDK];
  __shared__ __align__(16) __bf16 lB[128 * LDK];
  const int m0 = blockIdx.x * 128, n0 = blockIdx.y * 128;
  const int tid = threadIdx.x, lane = tid & 31, wave = tid >> 5;

  v8f acc[8];
#pragma unroll
  for (int f = 0; f < 8; ++f)
#pragma unroll
    for (int v = 0; v < 8; ++v) acc[f][v] = 0.0f;

  for (int kt = 0; kt < 1536; kt += 32) {
    // K-region is uniform per 32-wide tile (boundaries at 512/1024).
    if (kt < 512) {
      for (int idx = tid; idx < 4096; idx += 256) {      // query: f32 convert
        int k = idx & 31, m = idx >> 5;
        lA[m * LDK + k] = (__bf16)query[(size_t)(m0 + m) * 512 + kt + k];
      }
    } else {
      const __bf16* srcp = (kt < 1024) ? scx : tcx;
      int kg0 = kt - ((kt < 1024) ? 512 : 1024);
      async_tile_bf16(srcp + (size_t)m0 * 512, 512, kg0, lA, tid);
    }
    for (int idx = tid; idx < 4096; idx += 256) {        // Wo: f32 convert
      int n = idx & 127, k = idx >> 7;
      lB[n * LDK + k] = (__bf16)Wo[(size_t)(kt + k) * 512 + n0 + n];
    }
    wait_async0();
    __syncthreads();
    gemm_step(lA, lB, lane, wave, acc);
    __syncthreads();
  }

  const int half = lane >> 4, nl = lane & 15;
#pragma unroll
  for (int f = 0; f < 8; ++f)
#pragma unroll
    for (int v = 0; v < 8; ++v) {
      int row = m0 + wave * 16 + v + 8 * half;
      int col = n0 + f * 16 + nl;
      out[(size_t)row * 512 + col] = acc[f][v] + bo[col];
    }
}

// ------------------------------------------------------------------ launch
extern "C" void kernel_launch(void* const* d_in, const int* in_sizes, int n_in,
                              void* d_out, int out_size, void* d_ws, size_t ws_size,
                              hipStream_t stream) {
  const float* query = (const float*)d_in[0];
  const float* src   = (const float*)d_in[1];
  const float* trg   = (const float*)d_in[2];
  const float* Wq    = (const float*)d_in[3];
  const float* bq    = (const float*)d_in[4];
  const float* Ws    = (const float*)d_in[5];
  const float* bs    = (const float*)d_in[6];
  const float* Wo    = (const float*)d_in[7];
  const float* bo    = (const float*)d_in[8];
  float* out = (float*)d_out;

  // Workspace layout (~13.6 MB)
  char* w = (char*)d_ws;
  float*  qF  = (float*)w;  w += 2048ull * 512 * sizeof(float);   // q proj (f32)
  __bf16* sB  = (__bf16*)w; w += 2048ull * 512 * sizeof(__bf16);  // src keys
  __bf16* tB  = (__bf16*)w; w += 2048ull * 512 * sizeof(__bf16);  // trg keys
  __bf16* wsB = (__bf16*)w; w += 2048ull * 128 * sizeof(__bf16);  // sum_t softmax
  __bf16* wtB = (__bf16*)w; w += 2048ull * 128 * sizeof(__bf16);  // sum_s softmax
  __bf16* scx = (__bf16*)w; w += 2048ull * 512 * sizeof(__bf16);  // s_ctx
  __bf16* tcx = (__bf16*)w; w += 2048ull * 512 * sizeof(__bf16);  // t_ctx

  dim3 blk(256);
  da_proj_kernel<<<dim3(16, 4), blk, 0, stream>>>(query, Wq, bq, qF, nullptr,
                                                  2048, 512, 512);
  da_proj_kernel<<<dim3(16, 4), blk, 0, stream>>>(src, Ws, bs, nullptr, sB,
                                                  2048, 512, 512);
  da_proj_kernel<<<dim3(16, 4), blk, 0, stream>>>(trg, Ws, bs, nullptr, tB,
                                                  2048, 512, 512);
  da_score_kernel<<<dim3(2048), blk, 0, stream>>>(qF, sB, tB, wsB, wtB);
  da_ctx_kernel<<<dim3(4, 16, 2), blk, 0, stream>>>(wsB, wtB, src, trg, scx, tcx);
  da_final_kernel<<<dim3(16, 4), blk, 0, stream>>>(query, scx, tcx, Wo, bo, out);
}
